// GraphAttentionLayer_4294967296183
// MI455X (gfx1250) — matrix operand order repaired and checked by reading, not compiled, not run on hardware
//
#include <hip/hip_runtime.h>
#include <math.h>

// GAT layer for MI455X (gfx1250): fused f16-WMMA neighbor GEMM + masked softmax.
// Shapes: B=16384, N=64, F_IN=128, H=4, D=32 (H*D=128).

typedef _Float16 half8   __attribute__((ext_vector_type(8)));
typedef _Float16 half16  __attribute__((ext_vector_type(16)));
typedef float    floatx8 __attribute__((ext_vector_type(8)));

#define B_TOT 16384
#define NNB   64
#define FIN   128
#define HD    128   // H*D
#define NH    4
#define DD    32
#define LK    136   // padded LDS stride in halves: 272B rows -> 4-dword bank
                    // rotation per row -> conflict-free ds_load_b128 fragments

struct __align__(16) Smem {
  _Float16 sWt[HD * LK];   // W transposed, f16: sWt[n*LK + k] = W[k][n]   34816 B
  _Float16 sX [32 * LK];   // neighbor chunk (32 rows), f16                 8704 B
  _Float16 sH [NNB * LK];  // h_nb = X @ W results, f16                    17408 B
  float    sNode[FIN];     //                                                512 B
  float    sA[2 * HD];     // a[h][0:32]=a1, a[h][32:64]=a2                 1024 B
  float    sTmp[HD];       //                                                512 B
  float    sE1[NH];
  float    sE[NNB * NH];
  float    sAlpha[NNB * NH];
  unsigned char sMask[NNB];
};                          // total ~65.1 KB (<= 64 KiB limit)

__global__ __launch_bounds__(128)
void gat_wmma_kernel(const float* __restrict__ node,
                     const float* __restrict__ nbr,
                     const unsigned char* __restrict__ mask,
                     const float* __restrict__ W,
                     const float* __restrict__ a,
                     float* __restrict__ out)
{
  __shared__ Smem s;
  const int tid  = threadIdx.x;
  const int b    = blockIdx.x;
  const int lane = tid & 31;
  const int wave = tid >> 5;
  const int hi   = lane >> 4;    // half-wave (selects K sub-block per ISA layout)
  const int l15  = lane & 15;

  // ---------------- Phase 0: stage W^T (f16), node, a, mask ----------------
  {
    // Thread tid owns COLUMN j=tid of W (= row j of W^T). For fixed k the
    // wave reads W[k*128 + (j..j+31)]: fully coalesced 128B, L2-resident.
    // Pack 8 halves -> one ds_store_b128: 16 LDS stores/thread (vs 128 b16).
    const int j = tid;
    #pragma unroll
    for (int k8 = 0; k8 < FIN; k8 += 8) {
      half8 pk;
      #pragma unroll
      for (int u = 0; u < 8; ++u)
        pk[u] = (_Float16)W[(size_t)(k8 + u) * HD + j];
      *reinterpret_cast<half8*>(&s.sWt[j * LK + k8]) = pk;
    }
  }
  s.sNode[tid]    = node[(size_t)b * FIN + tid];
  s.sA[tid]       = a[tid];
  s.sA[tid + 128] = a[tid + 128];
  if (tid < NNB) s.sMask[tid] = mask[(size_t)b * NNB + tid];
  __syncthreads();

  // ---------------- Phase 1: h_nb = X @ W via v_wmma_f32_16x16x32_f16 ------
  const int mwave = wave >> 1;         // local 16-row M tile within chunk
  const int nbase = (wave & 1) * 4;    // 4 N-tiles of 16 per wave
  for (int chunk = 0; chunk < 2; ++chunk) {
    { // load 32 neighbor rows -> f16 LDS; thread does a quarter-row (32 floats)
      int row = tid >> 2;
      int cb  = (tid & 3) * 32;
      const float4* xr = reinterpret_cast<const float4*>(
          nbr + (((size_t)b * NNB) + chunk * 32 + row) * FIN + cb);
      #pragma unroll
      for (int i = 0; i < 8; i += 2) {   // pack 2 float4 -> 1 ds_store_b128
        float4 v0 = xr[i];
        float4 v1 = xr[i + 1];
        half8 pk;
        pk[0] = (_Float16)v0.x; pk[1] = (_Float16)v0.y;
        pk[2] = (_Float16)v0.z; pk[3] = (_Float16)v0.w;
        pk[4] = (_Float16)v1.x; pk[5] = (_Float16)v1.y;
        pk[6] = (_Float16)v1.z; pk[7] = (_Float16)v1.w;
        *reinterpret_cast<half8*>(&s.sX[row * LK + cb + i * 4]) = pk;
      }
    }
    __syncthreads();

    floatx8 acc[4] = {};
    #pragma unroll
    for (int k0 = 0; k0 < FIN; k0 += 32) {
      // A fragment (16x32 f16): lane row = mwave*16 + l15,
      // K = k0 + hi*8 + {0..7} and {16..23}  -> two contiguous 16B loads.
      union { half16 v; half8 h[2]; } af;
      const _Float16* pa = &s.sX[(mwave * 16 + l15) * LK + k0 + hi * 8];
      af.h[0] = *reinterpret_cast<const half8*>(pa);
      af.h[1] = *reinterpret_cast<const half8*>(pa + 16);
      #pragma unroll
      for (int t = 0; t < 4; ++t) {
        // B fragment (32x16 f16): lane col = n0 + l15,
        // K = k0 + hi*8 + {0..7} and {16..23}  -> two contiguous 16B loads (sWt is W^T).
        int n0 = (nbase + t) * 16;
        union { half16 v; half8 h[2]; } bf;
        const _Float16* pb = &s.sWt[(n0 + l15) * LK + k0 + hi * 8];
        bf.h[0] = *reinterpret_cast<const half8*>(pb);
        bf.h[1] = *reinterpret_cast<const half8*>(pb + 16);
        acc[t] = __builtin_amdgcn_wmma_f32_16x16x32_f16(
            false, af.v, false, bf.v, (short)0, acc[t], false, false);
      }
    }
    { // D layout: VGPR r -> row r (+8 for lanes 16..31), col = lane&15
      int gr = chunk * 32 + mwave * 16 + hi * 8;
      #pragma unroll
      for (int t = 0; t < 4; ++t) {
        int col = (nbase + t) * 16 + l15;
        #pragma unroll
        for (int r = 0; r < 8; ++r)
          s.sH[(gr + r) * LK + col] = (_Float16)acc[t][r];
      }
    }
    __syncthreads();
  }

  // ---------------- Phase 2: h_node = node @ W; e1[h] = a1[h] . h_node[h] --
  {
    float hn = 0.f;
    const _Float16* wrow = &s.sWt[tid * LK];   // row tid of W^T = column tid of W
    #pragma unroll 8
    for (int k = 0; k < FIN; ++k) hn += s.sNode[k] * (float)wrow[k];
    int h = tid >> 5, d = tid & 31;
    s.sTmp[tid] = hn * s.sA[h * 64 + d];       // a1[h][d]
  }
  __syncthreads();
  if (tid < NH) {
    float e1 = 0.f;
    #pragma unroll
    for (int i = 0; i < DD; ++i) e1 += s.sTmp[tid * DD + i];
    s.sE1[tid] = e1;
  }
  __syncthreads();

  // ---------------- Phase 3: e[n][h] = leaky_relu(e1 + a2.h_nb), mask ------
  for (int p = tid; p < NNB * NH; p += 128) {
    int n = p >> 2, h = p & 3;
    float e2 = 0.f;
    const _Float16* hr = &s.sH[n * LK + h * DD];
    const float*    a2 = &s.sA[h * 64 + 32];
    #pragma unroll
    for (int d = 0; d < DD; ++d) e2 += (float)hr[d] * a2[d];
    float e = s.sE1[h] + e2;
    e = (e > 0.f) ? e : 0.2f * e;
    if (!s.sMask[n]) e = -__builtin_inff();
    s.sE[n * NH + h] = e;
  }
  __syncthreads();

  // ---------------- Phase 4: softmax over n per head (wave w -> head w) ----
  {
    int h = wave;
    float ea = s.sE[lane * NH + h];
    float eb = s.sE[(lane + 32) * NH + h];
    float m = fmaxf(ea, eb);
    #pragma unroll
    for (int off = 16; off > 0; off >>= 1)
      m = fmaxf(m, __shfl_xor(m, off, 32));
    float xa = __expf(ea - m), xb = __expf(eb - m);
    float sum = xa + xb;
    #pragma unroll
    for (int off = 16; off > 0; off >>= 1)
      sum += __shfl_xor(sum, off, 32);
    float inv = 1.f / sum;
    s.sAlpha[lane * NH + h]        = xa * inv;
    s.sAlpha[(lane + 32) * NH + h] = xb * inv;
  }
  __syncthreads();

  // ---------------- Phase 5: out[b][c] = sum_n alpha[n][c/32] * h_nb[n][c] -
  {
    int c = tid, h = c >> 5;
    float o = 0.f;
    #pragma unroll 8
    for (int n = 0; n < NNB; ++n)
      o += s.sAlpha[n * NH + h] * (float)s.sH[n * LK + c];
    out[(size_t)b * HD + c] = o;
  }
}

extern "C" void kernel_launch(void* const* d_in, const int* in_sizes, int n_in,
                              void* d_out, int out_size, void* d_ws, size_t ws_size,
                              hipStream_t stream) {
  const float*         node = (const float*)d_in[0];
  const float*         nbr  = (const float*)d_in[1];
  const unsigned char* mask = (const unsigned char*)d_in[2];  // jnp.bool_ = 1 byte
  const float*         W    = (const float*)d_in[3];
  const float*         a    = (const float*)d_in[4];
  float*               out  = (float*)d_out;
  gat_wmma_kernel<<<B_TOT, 128, 0, stream>>>(node, nbr, mask, W, a, out);
}
